// Attn_68289980007139
// MI455X (gfx1250) — matrix operand order repaired and checked by reading, not compiled, not run on hardware
//
#include <hip/hip_runtime.h>
#include <hip/hip_bf16.h>
#include <math.h>

// Problem constants (match reference)
#define HIDDEN 512
#define ATTD   512
#define BB     32
#define SS     2048
#define M_TOT  (BB * SS)      // 65536 rows
#define KDIM   (2 * HIDDEN)   // 1024

typedef __attribute__((ext_vector_type(16))) _Float16 v16h;
typedef __attribute__((ext_vector_type(2)))  _Float16 h2;
typedef __attribute__((ext_vector_type(8)))  float    v8f;

// ---------------------------------------------------------------------------
// Kernel 0a: decoder_att[b][a] = dec_hidden[b] . W_dec[:,a] + b_dec[a]
// 16384 outputs, trivial (17 MFLOP)
// ---------------------------------------------------------------------------
__global__ __launch_bounds__(256) void k_dec_att(const float* __restrict__ dh,
                                                 const float* __restrict__ Wd,
                                                 const float* __restrict__ bd,
                                                 float* __restrict__ dec_att) {
  int idx = blockIdx.x * 256 + threadIdx.x;   // [0, 16384)
  int b = idx >> 9, a = idx & 511;
  float acc = bd[a];
  const float* h = dh + b * HIDDEN;
  for (int k = 0; k < HIDDEN; ++k) acc = fmaf(h[k], Wd[k * ATTD + a], acc);
  dec_att[idx] = acc;
}

// ---------------------------------------------------------------------------
// Kernel 0b: Wt[n][k] = (f16) W_enc[k][n]   (transposed f16 copy of W_enc,
// so each lane's B fragment is a contiguous 32-byte run along K)
// ---------------------------------------------------------------------------
__global__ __launch_bounds__(256) void k_transposeW(const float* __restrict__ We,
                                                    _Float16* __restrict__ Wt) {
  int idx = blockIdx.x * 256 + threadIdx.x;   // [0, 1024*512)
  int n = idx & 511, k = idx >> 9;            // coalesced read of We[idx]
  Wt[(size_t)n * KDIM + k] = (_Float16)We[idx];
}

// ---------------------------------------------------------------------------
// Kernel 1: fused GEMM (encoder_att) + bias + tanh-score + mask.
// Block = 256 threads (8 waves, 2(M) x 4(N) wave grid), tile = 64 x 512.
// Each wave: 2 M-subtiles x 8 N-subtiles of 16x16, f32 accumulators,
// v_wmma_f32_16x16x32_f16 over K in steps of 32.
//
// A fragment (16x32 f16), K-contiguous like B: lanes 0-15 hold M=lane, K in
// 0..15; lanes 16-31 hold M=lane-16, K in 16..31;
// register halves ascend within each 8-run.
// B fragment (32x16 f16): lanes 0-15 hold N=lane, K=0..15 ascending;
// lanes 16-31 hold N=lane-16, K=16..31 (loaded from K-contiguous Wt rows).
// C/D layout: lane = N (mod 16), VGPR v = M row v (lanes<16) / v+8 (lanes>=16).
// ---------------------------------------------------------------------------
__global__ __launch_bounds__(256) void k_gemm_score(
    const float*   __restrict__ enc,      // [M_TOT][KDIM] f32
    const _Float16* __restrict__ Wt,      // [ATTD][KDIM] f16 (transposed W_enc)
    const float*   __restrict__ benc,     // [ATTD]
    const float*   __restrict__ dec_att,  // [BB][ATTD]
    const float*   __restrict__ Watt,     // [ATTD]
    const float*   __restrict__ batt,     // [1]
    const unsigned char* __restrict__ masks, // [BB*SS] (jax bool = 1 byte)
    _Float16*      __restrict__ encAtt,   // [M_TOT][ATTD] f16 out
    float*         __restrict__ scores)   // [M_TOT] out
{
  const int tid    = threadIdx.x;
  const int lane   = tid & 31;
  const int wid    = tid >> 5;
  const int wave_m = wid >> 2;       // 0..1
  const int wave_n = wid & 3;        // 0..3
  const int lh     = lane >> 4;      // lane half
  const int l15    = lane & 15;

  const int m0    = blockIdx.x * 64;        // 64 | 2048 -> never crosses batch
  const int mwave = m0 + wave_m * 32;
  const int nwave = wave_n * 128;
  const int bidx  = m0 >> 11;               // batch index, uniform per block

  v8f zero = {};
  v8f acc[2][8];
  #pragma unroll
  for (int i = 0; i < 2; ++i)
    #pragma unroll
    for (int j = 0; j < 8; ++j) acc[i][j] = zero;

  for (int k0 = 0; k0 < KDIM; k0 += 32) {
    // --- A fragments: global f32 -> f16 registers ---
    v16h afr[2];
    #pragma unroll
    for (int ms = 0; ms < 2; ++ms) {
      const float* ap = enc + (size_t)(mwave + ms * 16 + l15) * KDIM + k0 + lh * 16;
      float4 r0 = *(const float4*)(ap);
      float4 r1 = *(const float4*)(ap + 4);
      float4 r2 = *(const float4*)(ap + 8);    // contiguous K-run (k = 16h + s)
      float4 r3 = *(const float4*)(ap + 12);
      v16h t;
      t[0]=(_Float16)r0.x;  t[1]=(_Float16)r0.y;  t[2]=(_Float16)r0.z;  t[3]=(_Float16)r0.w;
      t[4]=(_Float16)r1.x;  t[5]=(_Float16)r1.y;  t[6]=(_Float16)r1.z;  t[7]=(_Float16)r1.w;
      t[8]=(_Float16)r2.x;  t[9]=(_Float16)r2.y;  t[10]=(_Float16)r2.z; t[11]=(_Float16)r2.w;
      t[12]=(_Float16)r3.x; t[13]=(_Float16)r3.y; t[14]=(_Float16)r3.z; t[15]=(_Float16)r3.w;
      afr[ms] = t;
    }
    // --- B fragments: 32 contiguous bytes per lane from Wt ---
    v16h bfr[8];
    #pragma unroll
    for (int ns = 0; ns < 8; ++ns) {
      const _Float16* bp = Wt + (size_t)(nwave + ns * 16 + l15) * KDIM + k0 + lh * 16;
      bfr[ns] = *(const v16h*)bp;
    }
    // --- 16 WMMAs per K-step ---
    #pragma unroll
    for (int ms = 0; ms < 2; ++ms)
      #pragma unroll
      for (int ns = 0; ns < 8; ++ns)
        acc[ms][ns] = __builtin_amdgcn_wmma_f32_16x16x32_f16(
            false, afr[ms], false, bfr[ns], (short)0, acc[ms][ns], false, false);
  }

  // ---- epilogue: bias, store encoder_att (f16), tanh-score partials ----
  float scp[2][8] = {};
  #pragma unroll
  for (int ns = 0; ns < 8; ++ns) {
    const int   n  = nwave + ns * 16 + l15;
    const float be = benc[n];
    const float dv = dec_att[bidx * ATTD + n];
    const float wv = Watt[n];
    #pragma unroll
    for (int ms = 0; ms < 2; ++ms) {
      v8f c = acc[ms][ns];
      #pragma unroll
      for (int v = 0; v < 8; ++v) {
        float e = c[v] + be;                             // encoder_att value
        int m = mwave + ms * 16 + v + lh * 8;
        encAtt[(size_t)m * ATTD + n] = (_Float16)e;
        scp[ms][v] = fmaf(tanhf(e + dv), wv, scp[ms][v]);
      }
    }
  }

  // reduce score partials: across the 16 lanes of each half, then across the
  // 4 N-waves via LDS in a fixed order (deterministic, no float atomics).
  __shared__ float ldsS[4][64];
  #pragma unroll
  for (int ms = 0; ms < 2; ++ms)
    #pragma unroll
    for (int v = 0; v < 8; ++v) {
      float s = scp[ms][v];
      s += __shfl_xor(s, 1);
      s += __shfl_xor(s, 2);
      s += __shfl_xor(s, 4);
      s += __shfl_xor(s, 8);           // stays inside each 16-lane half
      if (l15 == 0)
        ldsS[wave_n][wave_m * 32 + ms * 16 + v + lh * 8] = s;
    }
  __syncthreads();
  if (tid < 64) {
    float s = ldsS[0][tid] + ldsS[1][tid] + ldsS[2][tid] + ldsS[3][tid] + batt[0];
    int m = m0 + tid;
    if (masks[m]) s = -INFINITY;
    scores[m] = s;
  }
}

// ---------------------------------------------------------------------------
// Kernel 2a: softmax over S per batch (deterministic tree reductions)
// ---------------------------------------------------------------------------
__global__ __launch_bounds__(256) void k_softmax(const float* __restrict__ scores,
                                                 float* __restrict__ w) {
  const int b = blockIdx.x, tid = threadIdx.x;
  __shared__ float red[256];
  float v[8];
  float mx = -INFINITY;
  #pragma unroll
  for (int i = 0; i < 8; ++i) {
    v[i] = scores[b * SS + tid + i * 256];
    mx = fmaxf(mx, v[i]);
  }
  red[tid] = mx; __syncthreads();
  for (int o = 128; o > 0; o >>= 1) {
    if (tid < o) red[tid] = fmaxf(red[tid], red[tid + o]);
    __syncthreads();
  }
  const float M = red[0];
  __syncthreads();
  float sm = 0.f;
  #pragma unroll
  for (int i = 0; i < 8; ++i) { v[i] = expf(v[i] - M); sm += v[i]; }
  red[tid] = sm; __syncthreads();
  for (int o = 128; o > 0; o >>= 1) {
    if (tid < o) red[tid] += red[tid + o];
    __syncthreads();
  }
  const float inv = 1.0f / red[0];
  #pragma unroll
  for (int i = 0; i < 8; ++i) w[b * SS + tid + i * 256] = v[i] * inv;
}

// ---------------------------------------------------------------------------
// Kernel 2b: per-(batch, s-chunk) partial context sums. 512 blocks.
// Skips weight==0 rows (masked) -> ~halves the encAtt re-read.
// ---------------------------------------------------------------------------
__global__ __launch_bounds__(256) void k_ctx_partial(const _Float16* __restrict__ encAtt,
                                                     const float* __restrict__ w,
                                                     float* __restrict__ part) {
  const int b = blockIdx.x >> 4;
  const int chunk = blockIdx.x & 15;
  const int a0 = threadIdx.x * 2;
  const int srow0 = b * SS + chunk * 128;
  float ax = 0.f, ay = 0.f;
  for (int i = 0; i < 128; ++i) {
    const float wt = w[srow0 + i];         // uniform across the block
    if (wt != 0.0f) {
      h2 p = *(const h2*)(encAtt + (size_t)(srow0 + i) * ATTD + a0);
      ax = fmaf(wt, (float)p.x, ax);
      ay = fmaf(wt, (float)p.y, ay);
    }
  }
  float* o = part + (size_t)blockIdx.x * ATTD + a0;
  o[0] = ax; o[1] = ay;
}

// ---------------------------------------------------------------------------
// Kernel 2c: reduce the 16 chunks per batch (fixed order -> deterministic)
// ---------------------------------------------------------------------------
__global__ __launch_bounds__(256) void k_ctx_reduce(const float* __restrict__ part,
                                                    float* __restrict__ out) {
  int idx = blockIdx.x * 256 + threadIdx.x;   // [0, 16384)
  int b = idx >> 9, a = idx & 511;
  float s = 0.f;
  #pragma unroll
  for (int c = 0; c < 16; ++c) s += part[(size_t)(b * 16 + c) * ATTD + a];
  out[idx] = s;
}

// ---------------------------------------------------------------------------
extern "C" void kernel_launch(void* const* d_in, const int* in_sizes, int n_in,
                              void* d_out, int out_size, void* d_ws, size_t ws_size,
                              hipStream_t stream) {
  const float* enc  = (const float*)d_in[0];          // [32,2048,1024]
  const float* dh   = (const float*)d_in[1];          // [32,512]
  const unsigned char* masks = (const unsigned char*)d_in[2]; // [32,2048,1] bool
  const float* We   = (const float*)d_in[3];          // [1024,512]
  const float* be   = (const float*)d_in[4];          // [512]
  const float* Wd   = (const float*)d_in[5];          // [512,512]
  const float* bd   = (const float*)d_in[6];          // [512]
  const float* Wa   = (const float*)d_in[7];          // [512,1]
  const float* ba   = (const float*)d_in[8];          // [1]
  float* out = (float*)d_out;                         // [32,512] f32

  // workspace layout (bytes)
  char* ws = (char*)d_ws;
  _Float16* encAtt  = (_Float16*)(ws);                    // 65536*512*2 = 64 MiB
  _Float16* Wt      = (_Float16*)(ws + 67108864);         // 512*1024*2  =  1 MiB
  float*    dec_att = (float*)   (ws + 68157440);         // 16384*4     = 64 KiB
  float*    scores  = (float*)   (ws + 68222976);         // 65536*4     = 256 KiB
  float*    wsoft   = (float*)   (ws + 68485120);         // 65536*4     = 256 KiB
  float*    part    = (float*)   (ws + 68747264);         // 512*512*4   =  1 MiB

  k_dec_att   <<<  64, 256, 0, stream>>>(dh, Wd, bd, dec_att);
  k_transposeW<<<2048, 256, 0, stream>>>(We, Wt);
  k_gemm_score<<<1024, 256, 0, stream>>>(enc, Wt, be, dec_att, Wa, ba, masks,
                                         encAtt, scores);
  k_softmax   <<<  32, 256, 0, stream>>>(scores, wsoft);
  k_ctx_partial<<< 512, 256, 0, stream>>>(encAtt, wsoft, part);
  k_ctx_reduce<<<  64, 256, 0, stream>>>(part, out);
}
